// AnomalyTransformer_78554951844027
// MI455X (gfx1250) — matrix operand order repaired and checked by reading, not compiled, not run on hardware
//
#include <hip/hip_runtime.h>

// =====================================================================
// Anomaly-Transformer forward for MI455X (gfx1250), bf16 WMMA GEMM core.
// GEMM: 128x128 block tile, 8 wave32s (4x2), wave tile 32x64
//       => 8 v_wmma_f32_16x16x32_bf16 per K-step, LDS double-buffered,
//       both A and B fragments read as contiguous ds_load_b128.
// =====================================================================

typedef __attribute__((ext_vector_type(16))) __bf16 v16bf;
typedef __attribute__((ext_vector_type(8)))  float  v8f;
typedef __attribute__((ext_vector_type(4)))  __bf16 bf16x4;
typedef __bf16 bf16_t;

#define BM 128
#define BN 128
#define BKK 32
#define LDK (BKK + 8)          // padded k-stride for both As[m][k] and Bs[n][k]
#define TB 256

// ---------------- fragment loaders (this code's contiguous k order, k = 16*half + e, for A and B) -----
// A frag (16x32 bf16): lane 0-15 row M=lane, K={0..15};
//                      lane 16-31 row M=lane-16, K={16..31}.
// Stored As[m][k] (k-contiguous) -> per lane: 2 contiguous 16B runs.

__device__ __forceinline__ v16bf load_frag_a(const bf16_t* As, int rowBase, int lane) {
  const int half_id = lane >> 4;
  const int r = rowBase + (lane & 15);
  const bf16_t* p0 = As + r * LDK + half_id * 16;       // K = half*16 + 0..7
  const bf16_t* p1 = p0 + 8;                            // K = half*16 + 8..15
  v16bf a;
#pragma unroll
  for (int e = 0; e < 8; ++e) a[e] = p0[e];
#pragma unroll
  for (int e = 0; e < 8; ++e) a[8 + e] = p1[e];
  return a;
}

// B frag (32x16 bf16): lane -> n = lane&15, kg = lane>>4; elems e -> K=kg*16+e.
// Stored Bs[n][k] (k-contiguous) -> one contiguous 32B run per lane.

__device__ __forceinline__ v16bf load_frag_b(const bf16_t* Bs, int colBase, int lane) {
  const int kg = lane >> 4;
  const int n = colBase + (lane & 15);
  const bf16_t* p = Bs + n * LDK + kg * 16;
  v16bf b;
#pragma unroll
  for (int e = 0; e < 16; ++e) b[e] = p[e];
  return b;
}

// ---------------- batched bf16 WMMA GEMM ------------------------------
// C[b,h] = A[b,h] (MxK) * B[b,h] (KxN, or NxK if transB) + bias
// Epilogue: optional per-column bias, optional exact GELU,
// writes fp32 (Cf) and/or bf16 (Cb). K must be a multiple of 32.

__global__ __launch_bounds__(TB)
void gemm_bf16_wmma(const bf16_t* __restrict__ A, const bf16_t* __restrict__ Bm,
                    const float* __restrict__ bias,
                    float* __restrict__ Cf, bf16_t* __restrict__ Cb,
                    int M, int N, int K, int lda, int ldb, int ldc,
                    long sAb, long sAh, long sBb, long sBh, long sCb, long sCh,
                    int Hdim, int transB, int act)
{
  __shared__ bf16_t As[2][BM * LDK];
  __shared__ bf16_t Bs[2][BN * LDK];

  const int z = blockIdx.z;
  const int bb = z / Hdim, hh = z - bb * Hdim;
  A  += (long)bb * sAb + (long)hh * sAh;
  Bm += (long)bb * sBb + (long)hh * sBh;
  const long coff = (long)bb * sCb + (long)hh * sCh;

  const int m0 = blockIdx.y * BM;
  const int n0 = blockIdx.x * BN;
  const int tid = threadIdx.x;
  const int lane = tid & 31;
  const int wave = tid >> 5;
  const int wm = wave >> 1;   // 0..3 -> row tile 32
  const int wn = wave & 1;    // 0..1 -> col tile 64

  const v8f vz = {0.f, 0.f, 0.f, 0.f, 0.f, 0.f, 0.f, 0.f};
  v8f acc[2][4];
#pragma unroll
  for (int i = 0; i < 2; ++i)
#pragma unroll
    for (int j = 0; j < 4; ++j) acc[i][j] = vz;

  const bf16_t bz = (bf16_t)0.0f;
  const bf16x4 zv = {bz, bz, bz, bz};

  bf16x4 aReg[4], bReg[4];

  // stage global tile (k0) into registers (with M/N guards)
  auto stage_load = [&](int k0) {
#pragma unroll
    for (int it = 0; it < 4; ++it) {
      const int v = tid + it * TB;
      const int row = v >> 3;                // 0..127
      const int c4 = (v & 7) << 2;           // 0..28
      const int gm = m0 + row;
      aReg[it] = zv;
      if (gm < M) aReg[it] = *(const bf16x4*)(A + (long)gm * lda + (k0 + c4));
    }
    if (!transB) {
      // global B is KxN: vector along n
#pragma unroll
      for (int it = 0; it < 4; ++it) {
        const int v = tid + it * TB;
        const int row = v >> 5;              // k: 0..31
        const int c4 = (v & 31) << 2;        // n: 0..124
        const int gn = n0 + c4;
        bReg[it] = zv;
        if (gn < N) bReg[it] = *(const bf16x4*)(Bm + (long)(k0 + row) * ldb + gn);
      }
    } else {
      // global B is NxK: vector along k
#pragma unroll
      for (int it = 0; it < 4; ++it) {
        const int v = tid + it * TB;
        const int nn = v >> 3;               // n: 0..127
        const int k4 = (v & 7) << 2;         // k: 0..28
        const int gn = n0 + nn;
        bReg[it] = zv;
        if (gn < N) bReg[it] = *(const bf16x4*)(Bm + (long)gn * ldb + (k0 + k4));
      }
    }
  };

  // store staged registers into LDS buffer `buf` (Bs stored n-major)
  auto stage_store = [&](int buf) {
#pragma unroll
    for (int it = 0; it < 4; ++it) {
      const int v = tid + it * TB;
      const int row = v >> 3;
      const int c4 = (v & 7) << 2;
      *(bf16x4*)(&As[buf][row * LDK + c4]) = aReg[it];
    }
    if (!transB) {
#pragma unroll
      for (int it = 0; it < 4; ++it) {
        const int v = tid + it * TB;
        const int row = v >> 5;              // k
        const int c4 = (v & 31) << 2;        // n
        Bs[buf][(c4 + 0) * LDK + row] = bReg[it][0];
        Bs[buf][(c4 + 1) * LDK + row] = bReg[it][1];
        Bs[buf][(c4 + 2) * LDK + row] = bReg[it][2];
        Bs[buf][(c4 + 3) * LDK + row] = bReg[it][3];
      }
    } else {
#pragma unroll
      for (int it = 0; it < 4; ++it) {
        const int v = tid + it * TB;
        const int nn = v >> 3;
        const int k4 = (v & 7) << 2;
        *(bf16x4*)(&Bs[buf][nn * LDK + k4]) = bReg[it];
      }
    }
  };

  const int numK = K / BKK;
  stage_load(0);
  stage_store(0);
  int cur = 0;

  for (int t = 0; t < numK; ++t) {
    __syncthreads();
    if (t + 1 < numK) stage_load((t + 1) * BKK);

    v16bf af0 = load_frag_a(As[cur], wm * 32,      lane);
    v16bf af1 = load_frag_a(As[cur], wm * 32 + 16, lane);
    v16bf bf0 = load_frag_b(Bs[cur], wn * 64,      lane);
    v16bf bf1 = load_frag_b(Bs[cur], wn * 64 + 16, lane);
    v16bf bf2 = load_frag_b(Bs[cur], wn * 64 + 32, lane);
    v16bf bf3 = load_frag_b(Bs[cur], wn * 64 + 48, lane);

    acc[0][0] = __builtin_amdgcn_wmma_f32_16x16x32_bf16(false, af0, false, bf0, (short)0, acc[0][0], false, false);
    acc[0][1] = __builtin_amdgcn_wmma_f32_16x16x32_bf16(false, af0, false, bf1, (short)0, acc[0][1], false, false);
    acc[0][2] = __builtin_amdgcn_wmma_f32_16x16x32_bf16(false, af0, false, bf2, (short)0, acc[0][2], false, false);
    acc[0][3] = __builtin_amdgcn_wmma_f32_16x16x32_bf16(false, af0, false, bf3, (short)0, acc[0][3], false, false);
    acc[1][0] = __builtin_amdgcn_wmma_f32_16x16x32_bf16(false, af1, false, bf0, (short)0, acc[1][0], false, false);
    acc[1][1] = __builtin_amdgcn_wmma_f32_16x16x32_bf16(false, af1, false, bf1, (short)0, acc[1][1], false, false);
    acc[1][2] = __builtin_amdgcn_wmma_f32_16x16x32_bf16(false, af1, false, bf2, (short)0, acc[1][2], false, false);
    acc[1][3] = __builtin_amdgcn_wmma_f32_16x16x32_bf16(false, af1, false, bf3, (short)0, acc[1][3], false, false);

    if (t + 1 < numK) {
      stage_store(1 - cur);   // safe: buffer (1-cur) last read before this iteration's barrier
      cur ^= 1;
    }
  }

  // ---- epilogue: C/D layout = VGPR r -> M=r (lanes 0-15) / M=r+8 ----
  const int nloc = lane & 15;
  const int half_id = lane >> 4;
#pragma unroll
  for (int i = 0; i < 2; ++i) {
#pragma unroll
    for (int j = 0; j < 4; ++j) {
      const int mb  = m0 + wm * 32 + i * 16 + half_id * 8;
      const int nb2 = n0 + wn * 64 + j * 16 + nloc;
#pragma unroll
      for (int r = 0; r < 8; ++r) {
        const int m = mb + r;
        if (m < M && nb2 < N) {
          float xv = acc[i][j][r];
          if (bias) xv += bias[nb2];
          if (act == 1) xv = 0.5f * xv * (1.0f + erff(xv * 0.70710678118654752f));
          const long o = coff + (long)m * ldc + nb2;
          if (Cf) Cf[o] = xv;
          if (Cb) Cb[o] = (bf16_t)xv;
        }
      }
    }
  }
}

// ---------------- layernorm over D=512, optional residual -------------

__global__ __launch_bounds__(256)
void ln_kernel(const float* __restrict__ X, const float* __restrict__ R,
               const float* __restrict__ g, const float* __restrict__ b,
               float* __restrict__ Yf, bf16_t* __restrict__ Yb, int Dm)
{
  __shared__ float red[256];
  const long row = blockIdx.x;
  const float* x = X + row * (long)Dm;
  const float* r = R ? (R + row * (long)Dm) : nullptr;
  const int tid = threadIdx.x;
  float lv[2];
  float s = 0.f;
#pragma unroll
  for (int it = 0; it < 2; ++it) {
    const int idx = tid + it * 256;
    float val = x[idx] + (r ? r[idx] : 0.f);
    lv[it] = val;
    s += val;
  }
  red[tid] = s; __syncthreads();
  for (int off = 128; off; off >>= 1) { if (tid < off) red[tid] += red[tid + off]; __syncthreads(); }
  const float mean = red[0] / (float)Dm;
  __syncthreads();
  float ss = 0.f;
#pragma unroll
  for (int it = 0; it < 2; ++it) { const float d = lv[it] - mean; ss += d * d; }
  red[tid] = ss; __syncthreads();
  for (int off = 128; off; off >>= 1) { if (tid < off) red[tid] += red[tid + off]; __syncthreads(); }
  const float inv = rsqrtf(red[0] / (float)Dm + 1e-5f);
#pragma unroll
  for (int it = 0; it < 2; ++it) {
    const int idx = tid + it * 256;
    const float y = g[idx] * (lv[it] - mean) * inv + b[idx];
    if (Yf) Yf[row * (long)Dm + idx] = y;
    if (Yb) Yb[row * (long)Dm + idx] = (bf16_t)y;
  }
}

// ---------------- row softmax (Lk<=512), writes bf16 + opt fp32 -------

__global__ __launch_bounds__(256)
void softmax_kernel(const float* __restrict__ S, bf16_t* __restrict__ Pb,
                    float* __restrict__ Pf, float scale, int Lk)
{
  __shared__ float red[256];
  const long row = blockIdx.x;
  const float* s = S + row * (long)Lk;
  const int tid = threadIdx.x;
  float lv[2];
  int cnt = 0;
  float mx = -3.4e38f;
  for (int j = tid; j < Lk; j += 256) { const float v = s[j] * scale; lv[cnt++] = v; mx = fmaxf(mx, v); }
  red[tid] = mx; __syncthreads();
  for (int off = 128; off; off >>= 1) { if (tid < off) red[tid] = fmaxf(red[tid], red[tid + off]); __syncthreads(); }
  mx = red[0]; __syncthreads();
  float sum = 0.f;
  for (int k = 0; k < cnt; ++k) { const float e = expf(lv[k] - mx); lv[k] = e; sum += e; }
  red[tid] = sum; __syncthreads();
  for (int off = 128; off; off >>= 1) { if (tid < off) red[tid] += red[tid + off]; __syncthreads(); }
  const float inv = 1.f / red[0];
  cnt = 0;
  for (int j = tid; j < Lk; j += 256) {
    const float p = lv[cnt++] * inv;
    Pb[row * (long)Lk + j] = (bf16_t)p;
    if (Pf) Pf[row * (long)Lk + j] = p;
  }
}

// ---------------- sigma: [B,L,D] @ Wsig[D,H] -> transformed [B,H,L] ---

__global__ __launch_bounds__(256)
void sigma_kernel(const float* __restrict__ Xo, const float* __restrict__ Wsig,
                  const float* __restrict__ bsig, float* __restrict__ out,
                  int L, int Dm, int Hh)
{
  const long rowIdx = blockIdx.x;           // b*L + l
  const long b = rowIdx / L, l = rowIdx % L;
  const int tid = threadIdx.x, lane = tid & 31, h = tid >> 5; // 8 waves -> heads
  const float* x = Xo + rowIdx * (long)Dm;
  float p = 0.f;
  for (int d = lane; d < Dm; d += 32) p += x[d] * Wsig[d * Hh + h];
  for (int off = 16; off; off >>= 1) p += __shfl_xor(p, off);
  if (lane == 0) {
    const float sv = p + bsig[h];
    const float sg = 1.f / (1.f + expf(-5.f * sv));
    const float sig = powf(3.f, sg + 1e-5f) - 1.f;
    out[(b * Hh + h) * (long)L + l] = sig;
  }
}

// ---------------- Gaussian prior from sigma ---------------------------

__global__ void prior_kernel(const float* __restrict__ sig, float* __restrict__ prior,
                             int L, long total)
{
  const long idx = (long)blockIdx.x * blockDim.x + threadIdx.x;
  if (idx >= total) return;
  const long srow = idx / L;
  const int scol = (int)(idx % L);
  const int l = (int)(srow % L);
  const float sg = sig[srow];
  const float d = fabsf((float)(l - scol));
  const float t = d / sg;
  prior[idx] = expf(-0.5f * t * t) * 0.3989422804014327f / sg;
}

// ---------------- positional embedding add ----------------------------

__global__ void add_pe_kernel(float* __restrict__ Xf, bf16_t* __restrict__ Xb,
                              int L, int Dm, long total)
{
  const long idx = (long)blockIdx.x * blockDim.x + threadIdx.x;
  if (idx >= total) return;
  const int d = (int)(idx % Dm);
  const long rl = idx / Dm;
  const int pos = (int)(rl % L);
  const int i2 = d & ~1;
  const float ang = (float)pos * expf(-((float)i2 / (float)Dm) * 9.210340371976184f);
  const float pe = (d & 1) ? cosf(ang) : sinf(ang);
  const float v = Xf[idx] + pe;
  Xf[idx] = v;
  Xb[idx] = (bf16_t)v;
}

// ---------------- fp32 -> bf16 convert --------------------------------

__global__ void cvt_kernel(const float* __restrict__ in, bf16_t* __restrict__ out, long n)
{
  const long idx = (long)blockIdx.x * blockDim.x + threadIdx.x;
  if (idx < n) out[idx] = (bf16_t)in[idx];
}

// ---------------- transpose [B,W,C] -> bf16 [B,C,W] -------------------

__global__ void transpose_cvt_kernel(const float* __restrict__ x, bf16_t* __restrict__ xt,
                                     int Wn, int Cn, long total)
{
  const long idx = (long)blockIdx.x * blockDim.x + threadIdx.x;
  if (idx >= total) return;
  const long b = idx / ((long)Cn * Wn);
  const long rem = idx % ((long)Cn * Wn);
  const int c = (int)(rem / Wn);
  const int w = (int)(rem % Wn);
  xt[idx] = (bf16_t)x[b * (long)Wn * Cn + (long)w * Cn + c];
}

// ---------------- feature row norms -----------------------------------

__global__ __launch_bounds__(256)
void rownorm_kernel(const float* __restrict__ F, float* __restrict__ nrm, int Dm)
{
  __shared__ float red[256];
  const long row = blockIdx.x;
  const int tid = threadIdx.x;
  float s = 0.f;
  for (int d = tid; d < Dm; d += 256) { const float v = F[row * (long)Dm + d]; s += v * v; }
  red[tid] = s; __syncthreads();
  for (int off = 128; off; off >>= 1) { if (tid < off) red[tid] += red[tid + off]; __syncthreads(); }
  if (tid == 0) nrm[row] = sqrtf(red[0]);
}

// ---------------- contrastive logits ----------------------------------

__global__ __launch_bounds__(256)
void contrastive_kernel(const float* __restrict__ X, const float* __restrict__ F,
                        const float* __restrict__ fn, const int* __restrict__ c,
                        float* __restrict__ out, int Dm, int Kc, float invT)
{
  __shared__ float red[256];
  __shared__ float sim[64];
  __shared__ float xn;
  const long row = blockIdx.x;
  const float* x = X + row * (long)Dm;
  const int tid = threadIdx.x, lane = tid & 31, wave = tid >> 5;
  float s = 0.f;
  for (int d = tid; d < Dm; d += 256) { const float v = x[d]; s += v * v; }
  red[tid] = s; __syncthreads();
  for (int off = 128; off; off >>= 1) { if (tid < off) red[tid] += red[tid + off]; __syncthreads(); }
  if (tid == 0) xn = sqrtf(red[0]);
  __syncthreads();
  for (int kk = wave; kk < Kc; kk += 8) {
    float p = 0.f;
    for (int d = lane; d < Dm; d += 32) p += x[d] * F[(long)kk * Dm + d];
    for (int off = 16; off; off >>= 1) p += __shfl_xor(p, off);
    if (lane == 0) sim[kk] = p / (xn * fn[kk]);
  }
  __syncthreads();
  const int cp = c[row];
  if (tid == 0) out[row * (long)Kc] = sim[cp] * invT;
  else if (tid < Kc) {
    const int j = tid - 1;
    const int col = j + (j >= cp ? 1 : 0);
    out[row * (long)Kc + tid] = sim[col] * invT;
  }
}

// ---------------- final enc combine -----------------------------------

__global__ void enc_combine_kernel(const float* __restrict__ e1, const float* __restrict__ e2,
                                   const float* __restrict__ e3,
                                   const float* __restrict__ bpt, const float* __restrict__ bpc,
                                   const float* __restrict__ bpo,
                                   float* __restrict__ out, int L, int Co, long total)
{
  const long idx = (long)blockIdx.x * blockDim.x + threadIdx.x;
  if (idx >= total) return;
  const long b = idx / ((long)L * Co);
  const long rem = idx % ((long)L * Co);
  const int l = (int)(rem / Co);
  const int co = (int)(rem % Co);
  const float v1 = e1[idx] + bpt[co];
  const float v2 = e2[(b * Co + co) * (long)L + l] + bpc[l];
  const float v3 = e3[idx] + bpo[co];
  out[idx] = (v1 + v2 + v3) * (1.0f / 3.0f);
}

// =====================================================================
// Host orchestration
// =====================================================================

extern "C" void kernel_launch(void* const* d_in, const int* in_sizes, int n_in,
                              void* d_out, int out_size, void* d_ws, size_t ws_size,
                              hipStream_t stream)
{
  (void)in_sizes; (void)n_in; (void)out_size; (void)ws_size;

  const int Bc = 8, WINc = 512, CINc = 128, COUTc = 128;
  const int Dc = 512, Hc = 8, DFc = 2048, ELc = 3, Kc = 64, Ec = 64;
  const int Nt = Bc * WINc;   // 4096
  const int Nc = Bc * CINc;   // 1024

  const float* x      = (const float*)d_in[0];
  const float* Wemb_t = (const float*)d_in[1];
  const float* bemb_t = (const float*)d_in[2];
  const float* Wemb_c = (const float*)d_in[3];
  const float* bemb_c = (const float*)d_in[4];
  const float* Wt = (const float*)d_in[5];
  const float* Wc = (const float*)d_in[6];
  const float* Wo = (const float*)d_in[7];
  const float* Wq = (const float*)d_in[8];
  const float* Wk = (const float*)d_in[9];
  const float* Wv = (const float*)d_in[10];
  const float* Wout = (const float*)d_in[11];
  const float* bt = (const float*)d_in[12];
  const float* bc = (const float*)d_in[13];
  const float* bo = (const float*)d_in[14];
  const float* bq = (const float*)d_in[15];
  const float* bk = (const float*)d_in[16];
  const float* bv = (const float*)d_in[17];
  const float* bout = (const float*)d_in[18];
  const float* Wsig = (const float*)d_in[19];
  const float* bsig = (const float*)d_in[20];
  const float* W1 = (const float*)d_in[21];
  const float* W3 = (const float*)d_in[22];
  const float* W5 = (const float*)d_in[23];
  const float* W2 = (const float*)d_in[24];
  const float* W4 = (const float*)d_in[25];
  const float* W6 = (const float*)d_in[26];
  const float* b1 = (const float*)d_in[27];
  const float* b3 = (const float*)d_in[28];
  const float* b5 = (const float*)d_in[29];
  const float* b2 = (const float*)d_in[30];
  const float* b4 = (const float*)d_in[31];
  const float* b6 = (const float*)d_in[32];
  const float* ln_g = (const float*)d_in[33];
  const float* ln_b = (const float*)d_in[34];
  const float* gn_g = (const float*)d_in[35];
  const float* gn_b = (const float*)d_in[36];
  const float* Wpt = (const float*)d_in[37];
  const float* bpt = (const float*)d_in[38];
  const float* Wpc = (const float*)d_in[39];
  const float* bpc = (const float*)d_in[40];
  const float* Wpo = (const float*)d_in[41];
  const float* bpo = (const float*)d_in[42];
  const float* feats_t = (const float*)d_in[43];
  const float* feats_c = (const float*)d_in[44];
  const float* feats_o = (const float*)d_in[45];
  const int* c_t = (const int*)d_in[46];
  const int* c_c = (const int*)d_in[47];
  const int* c_o = (const int*)d_in[48];

  float* outf = (float*)d_out;

  // ---- output slices (float counts) ----
  const long ENC_N   = (long)Bc * WINc * COUTc;                 // 524288
  const long SER_L   = (long)Bc * Hc * WINc * WINc;             // per layer
  const long SIG_L   = (long)Bc * Hc * WINc;                    // per layer
  const long series_off = ENC_N;
  const long prior_off  = series_off + (long)ELc * SER_L;
  const long sigma_off  = prior_off + (long)ELc * SER_L;
  const long lt_off     = sigma_off + (long)ELc * SIG_L;
  const long lc_off     = lt_off + (long)Nt * Kc;
  const long lo_off     = lc_off + (long)Nc * Kc;

  // ---- workspace bump allocator ----
  char* wsp = (char*)d_ws;
  size_t cur = 0;
  auto alloc = [&](size_t elems, size_t esz) -> void* {
    void* p = wsp + cur;
    cur += (elems * esz + 255) & ~(size_t)255;
    return p;
  };
  auto allocb = [&](size_t elems) { return (bf16_t*)alloc(elems, 2); };
  auto allocf = [&](size_t elems) { return (float*)alloc(elems, 4); };

  // bf16 weight copies
  bf16_t* wembt_b = allocb((size_t)CINc * Dc);
  bf16_t* wembc_b = allocb((size_t)WINc * Dc);
  bf16_t* Wt_b   = allocb((size_t)ELc * Dc * Dc);
  bf16_t* Wc_b   = allocb((size_t)ELc * Dc * Dc);
  bf16_t* Wo_b   = allocb((size_t)ELc * Dc * Dc);
  bf16_t* Wq_b   = allocb((size_t)ELc * Dc * Dc);
  bf16_t* Wk_b   = allocb((size_t)ELc * Dc * Dc);
  bf16_t* Wv_b   = allocb((size_t)ELc * Dc * Dc);
  bf16_t* Wout_b = allocb((size_t)ELc * Dc * Dc);
  bf16_t* W1_b = allocb((size_t)ELc * Dc * DFc);
  bf16_t* W3_b = allocb((size_t)ELc * Dc * DFc);
  bf16_t* W5_b = allocb((size_t)ELc * Dc * DFc);
  bf16_t* W2_b = allocb((size_t)ELc * DFc * Dc);
  bf16_t* W4_b = allocb((size_t)ELc * DFc * Dc);
  bf16_t* W6_b = allocb((size_t)ELc * DFc * Dc);
  bf16_t* Wpt_b = allocb((size_t)Dc * COUTc);
  bf16_t* Wpc_b = allocb((size_t)Dc * WINc);
  bf16_t* Wpo_b = allocb((size_t)Dc * COUTc);

  // activations
  bf16_t* x_bf  = allocb((size_t)Nt * CINc);
  bf16_t* xT_bf = allocb((size_t)Nc * WINc);
  float*  x_t_f = allocf((size_t)Nt * Dc); bf16_t* x_t_b = allocb((size_t)Nt * Dc);
  float*  x_c_f = allocf((size_t)Nc * Dc); bf16_t* x_c_b = allocb((size_t)Nc * Dc);
  float*  x_o_f = allocf((size_t)Nt * Dc); bf16_t* x_o_b = allocb((size_t)Nt * Dc);
  float*  xt_f  = allocf((size_t)Nt * Dc); bf16_t* xt_b  = allocb((size_t)Nt * Dc);
  float*  xc_f  = allocf((size_t)Nc * Dc); bf16_t* xc_b  = allocb((size_t)Nc * Dc);
  float*  xo_f  = allocf((size_t)Nt * Dc); bf16_t* xo_b  = allocb((size_t)Nt * Dc);
  bf16_t* q_b = allocb((size_t)Nt * Dc);
  bf16_t* k_b = allocb((size_t)Nt * Dc);
  bf16_t* v_b = allocb((size_t)Nt * Dc);
  float*  scores_f = allocf((size_t)Bc * Hc * WINc * WINc);
  bf16_t* ser_b    = allocb((size_t)Bc * Hc * WINc * WINc);
  bf16_t* att_b    = allocb((size_t)Nt * Dc);
  float*  attout_f = allocf((size_t)Nt * Dc);
  float*  x1_f = allocf((size_t)Nt * Dc); bf16_t* x1_b = allocb((size_t)Nt * Dc);
  bf16_t* hid_b = allocb((size_t)Nt * DFc);
  float*  y_f = allocf((size_t)Nt * Dc);
  bf16_t* ln_t_b = allocb((size_t)Nt * Dc);
  bf16_t* ln_c_b = allocb((size_t)Nc * Dc);
  bf16_t* ln_o_b = allocb((size_t)Nt * Dc);
  float* e1_f = allocf((size_t)Nt * COUTc);
  float* e2_f = allocf((size_t)Nc * WINc);
  float* e3_f = allocf((size_t)Nt * COUTc);
  float* fn_t = allocf(64);
  float* fn_c = allocf(64);
  float* fn_o = allocf(64);

  auto cvt = [&](const float* src, bf16_t* dst, long n) {
    cvt_kernel<<<dim3((unsigned)((n + 255) / 256)), dim3(256), 0, stream>>>(src, dst, n);
  };

  auto gemm = [&](const bf16_t* A, const bf16_t* B, const float* bias,
                  float* Cf, bf16_t* Cb, int M, int N, int K,
                  int lda, int ldb, int ldc,
                  long sAb, long sAh, long sBb, long sBh, long sCb, long sCh,
                  int Hdim, int nb, int transB, int act) {
    dim3 grid((N + BN - 1) / BN, (M + BM - 1) / BM, nb * Hdim);
    gemm_bf16_wmma<<<grid, dim3(TB), 0, stream>>>(A, B, bias, Cf, Cb, M, N, K,
                                                  lda, ldb, ldc, sAb, sAh, sBb, sBh,
                                                  sCb, sCh, Hdim, transB, act);
  };

  // ---- convert all weights to bf16 ----
  cvt(Wemb_t, wembt_b, (long)CINc * Dc);
  cvt(Wemb_c, wembc_b, (long)WINc * Dc);
  cvt(Wt, Wt_b, (long)ELc * Dc * Dc);
  cvt(Wc, Wc_b, (long)ELc * Dc * Dc);
  cvt(Wo, Wo_b, (long)ELc * Dc * Dc);
  cvt(Wq, Wq_b, (long)ELc * Dc * Dc);
  cvt(Wk, Wk_b, (long)ELc * Dc * Dc);
  cvt(Wv, Wv_b, (long)ELc * Dc * Dc);
  cvt(Wout, Wout_b, (long)ELc * Dc * Dc);
  cvt(W1, W1_b, (long)ELc * Dc * DFc);
  cvt(W3, W3_b, (long)ELc * Dc * DFc);
  cvt(W5, W5_b, (long)ELc * Dc * DFc);
  cvt(W2, W2_b, (long)ELc * DFc * Dc);
  cvt(W4, W4_b, (long)ELc * DFc * Dc);
  cvt(W6, W6_b, (long)ELc * DFc * Dc);
  cvt(Wpt, Wpt_b, (long)Dc * COUTc);
  cvt(Wpc, Wpc_b, (long)Dc * WINc);
  cvt(Wpo, Wpo_b, (long)Dc * COUTc);

  // ---- embeddings ----
  cvt(x, x_bf, (long)Nt * CINc);
  gemm(x_bf, wembt_b, bemb_t, x_t_f, nullptr, Nt, Dc, CINc,
       CINc, Dc, Dc, 0, 0, 0, 0, 0, 0, 1, 1, 0, 0);
  {
    const long tot = (long)Nt * Dc;
    add_pe_kernel<<<dim3((unsigned)((tot + 255) / 256)), dim3(256), 0, stream>>>(
        x_t_f, x_t_b, WINc, Dc, tot);
  }
  hipMemcpyAsync(x_o_f, x_t_f, (size_t)Nt * Dc * sizeof(float), hipMemcpyDeviceToDevice, stream);
  hipMemcpyAsync(x_o_b, x_t_b, (size_t)Nt * Dc * sizeof(bf16_t), hipMemcpyDeviceToDevice, stream);
  {
    const long tot = (long)Nc * WINc;
    transpose_cvt_kernel<<<dim3((unsigned)((tot + 255) / 256)), dim3(256), 0, stream>>>(
        x, xT_bf, WINc, CINc, tot);
  }
  gemm(xT_bf, wembc_b, bemb_c, x_c_f, x_c_b, Nc, Dc, WINc,
       WINc, Dc, Dc, 0, 0, 0, 0, 0, 0, 1, 1, 0, 0);

  // ---- encoder layers ----
  for (int i = 0; i < ELc; ++i) {
    const bf16_t* Wt_i = Wt_b + (long)i * Dc * Dc;
    const bf16_t* Wc_i = Wc_b + (long)i * Dc * Dc;
    const bf16_t* Wo_i = Wo_b + (long)i * Dc * Dc;
    const bf16_t* Wq_i = Wq_b + (long)i * Dc * Dc;
    const bf16_t* Wk_i = Wk_b + (long)i * Dc * Dc;
    const bf16_t* Wv_i = Wv_b + (long)i * Dc * Dc;
    const bf16_t* Wout_i = Wout_b + (long)i * Dc * Dc;
    const float* bt_i = bt + i * Dc;
    const float* bc_i = bc + i * Dc;
    const float* bo_i = bo + i * Dc;
    const float* bq_i = bq + i * Dc;
    const float* bk_i = bk + i * Dc;
    const float* bv_i = bv + i * Dc;
    const float* bout_i = bout + i * Dc;
    const float* ln_g_i = ln_g + i * Dc;
    const float* ln_b_i = ln_b + i * Dc;
    const float* Wsig_i = Wsig + (long)i * Dc * Hc;
    const float* bsig_i = bsig + i * Hc;
    const bf16_t* W1_i = W1_b + (long)i * Dc * DFc; const float* b1_i = b1 + i * DFc;
    const bf16_t* W3_i = W3_b + (long)i * Dc * DFc; const float* b3_i = b3 + i * DFc;
    const bf16_t* W5_i = W5_b + (long)i * Dc * DFc; const float* b5_i = b5 + i * DFc;
    const bf16_t* W2_i = W2_b + (long)i * DFc * Dc; const float* b2_i = b2 + i * Dc;
    const bf16_t* W4_i = W4_b + (long)i * DFc * Dc; const float* b4_i = b4 + i * Dc;
    const bf16_t* W6_i = W6_b + (long)i * DFc * Dc; const float* b6_i = b6 + i * Dc;

    // branch input projections
    gemm(x_t_b, Wt_i, bt_i, xt_f, xt_b, Nt, Dc, Dc, Dc, Dc, Dc, 0,0,0,0,0,0, 1, 1, 0, 0);
    gemm(x_c_b, Wc_i, bc_i, xc_f, xc_b, Nc, Dc, Dc, Dc, Dc, Dc, 0,0,0,0,0,0, 1, 1, 0, 0);
    gemm(x_o_b, Wo_i, bo_i, xo_f, xo_b, Nt, Dc, Dc, Dc, Dc, Dc, 0,0,0,0,0,0, 1, 1, 0, 0);

    // sigma (from projected xo) + prior for this layer
    float* sig_i = outf + sigma_off + (long)i * SIG_L;
    sigma_kernel<<<dim3((unsigned)Nt), dim3(256), 0, stream>>>(xo_f, Wsig_i, bsig_i,
                                                               sig_i, WINc, Dc, Hc);
    {
      const long tot = SER_L;
      prior_kernel<<<dim3((unsigned)((tot + 255) / 256)), dim3(256), 0, stream>>>(
          sig_i, outf + prior_off + (long)i * SER_L, WINc, tot);
    }

    auto run_branch = [&](const bf16_t* xqb, const float* xqf, const bf16_t* xkb,
                          int Lq, int Lk, float* seriesOut,
                          const bf16_t* Wf1, const float* bf1,
                          const bf16_t* Wf2, const float* bf2,
                          float* xdstF, bf16_t* xdstB) {
      const int Nq = Bc * Lq, Nk = Bc * Lk;
      gemm(xqb, Wq_i, bq_i, nullptr, q_b, Nq, Dc, Dc, Dc, Dc, Dc, 0,0,0,0,0,0, 1, 1, 0, 0);
      gemm(xkb, Wk_i, bk_i, nullptr, k_b, Nk, Dc, Dc, Dc, Dc, Dc, 0,0,0,0,0,0, 1, 1, 0, 0);
      gemm(xkb, Wv_i, bv_i, nullptr, v_b, Nk, Dc, Dc, Dc, Dc, Dc, 0,0,0,0,0,0, 1, 1, 0, 0);
      // scores[b,h] = q_bh (Lq x E) * k_bh^T (E x Lk)
      gemm(q_b, k_b, nullptr, scores_f, nullptr, Lq, Lk, Ec,
           Dc, Dc, Lk,
           (long)Lq * Dc, Ec, (long)Lk * Dc, Ec,
           (long)Hc * Lq * Lk, (long)Lq * Lk,
           Hc, Bc, 1, 0);
      softmax_kernel<<<dim3((unsigned)(Bc * Hc * Lq)), dim3(256), 0, stream>>>(
          scores_f, ser_b, seriesOut, 0.125f, Lk);
      // att[b,h] = series (Lq x Lk) * v_bh (Lk x E)
      gemm(ser_b, v_b, nullptr, nullptr, att_b, Lq, Ec, Lk,
           Lk, Dc, Dc,
           (long)Hc * Lq * Lk, (long)Lq * Lk, (long)Lk * Dc, Ec,
           (long)Lq * Dc, Ec,
           Hc, Bc, 0, 0);
      gemm(att_b, Wout_i, bout_i, attout_f, nullptr, Nq, Dc, Dc, Dc, Dc, Dc, 0,0,0,0,0,0, 1, 1, 0, 0);
      ln_kernel<<<dim3((unsigned)Nq), dim3(256), 0, stream>>>(xqf, attout_f, ln_g_i, ln_b_i,
                                                              x1_f, x1_b, Dc);
      gemm(x1_b, Wf1, bf1, nullptr, hid_b, Nq, DFc, Dc, Dc, DFc, DFc, 0,0,0,0,0,0, 1, 1, 0, 1);
      gemm(hid_b, Wf2, bf2, y_f, nullptr, Nq, Dc, DFc, DFc, Dc, Dc, 0,0,0,0,0,0, 1, 1, 0, 0);
      ln_kernel<<<dim3((unsigned)Nq), dim3(256), 0, stream>>>(x1_f, y_f, ln_g_i, ln_b_i,
                                                              xdstF, xdstB, Dc);
    };

    // t-branch: queries = temporal, keys/values = channel
    run_branch(xt_b, xt_f, xc_b, WINc, CINc, nullptr, W1_i, b1_i, W2_i, b2_i, x_t_f, x_t_b);
    // c-branch: queries = channel, keys/values = temporal
    run_branch(xc_b, xc_f, xt_b, CINc, WINc, nullptr, W3_i, b3_i, W4_i, b4_i, x_c_f, x_c_b);
    // o-branch: self attention, emits series to d_out
    run_branch(xo_b, xo_f, xo_b, WINc, WINc, outf + series_off + (long)i * SER_L,
               W5_i, b5_i, W6_i, b6_i, x_o_f, x_o_b);

    if (i == 2) {
      rownorm_kernel<<<dim3(64), dim3(256), 0, stream>>>(feats_t, fn_t, Dc);
      rownorm_kernel<<<dim3(64), dim3(256), 0, stream>>>(feats_c, fn_c, Dc);
      rownorm_kernel<<<dim3(64), dim3(256), 0, stream>>>(feats_o, fn_o, Dc);
      contrastive_kernel<<<dim3((unsigned)Nt), dim3(256), 0, stream>>>(
          x_t_f, feats_t, fn_t, c_t, outf + lt_off, Dc, Kc, 1.0f);
      contrastive_kernel<<<dim3((unsigned)Nc), dim3(256), 0, stream>>>(
          x_c_f, feats_c, fn_c, c_c, outf + lc_off, Dc, Kc, 1.0f);
      contrastive_kernel<<<dim3((unsigned)Nt), dim3(256), 0, stream>>>(
          x_o_f, feats_o, fn_o, c_o, outf + lo_off, Dc, Kc, 1.0f);
    }
  }

  // ---- projection heads ----
  ln_kernel<<<dim3((unsigned)Nt), dim3(256), 0, stream>>>(x_t_f, nullptr, gn_g, gn_b, nullptr, ln_t_b, Dc);
  ln_kernel<<<dim3((unsigned)Nc), dim3(256), 0, stream>>>(x_c_f, nullptr, gn_g, gn_b, nullptr, ln_c_b, Dc);
  ln_kernel<<<dim3((unsigned)Nt), dim3(256), 0, stream>>>(x_o_f, nullptr, gn_g, gn_b, nullptr, ln_o_b, Dc);
  gemm(ln_t_b, Wpt_b, nullptr, e1_f, nullptr, Nt, COUTc, Dc, Dc, COUTc, COUTc, 0,0,0,0,0,0, 1, 1, 0, 0);
  gemm(ln_c_b, Wpc_b, nullptr, e2_f, nullptr, Nc, WINc, Dc, Dc, WINc, WINc, 0,0,0,0,0,0, 1, 1, 0, 0);
  gemm(ln_o_b, Wpo_b, nullptr, e3_f, nullptr, Nt, COUTc, Dc, Dc, COUTc, COUTc, 0,0,0,0,0,0, 1, 1, 0, 0);
  {
    const long tot = ENC_N;
    enc_combine_kernel<<<dim3((unsigned)((tot + 255) / 256)), dim3(256), 0, stream>>>(
        e1_f, e2_f, e3_f, bpt, bpc, bpo, outf, WINc, COUTc, tot);
  }
}
